// GIN_77627238908071
// MI455X (gfx1250) — matrix operand order repaired and checked by reading, not compiled, not run on hardware
//
#include <hip/hip_runtime.h>
#include <hip/hip_bf16.h>
#include <math.h>

// ---------------------------------------------------------------------------
// Types for CDNA5 WMMA (wave32): 16x16x32 bf16 -> f32 accum
// ---------------------------------------------------------------------------
typedef __bf16 bf16_t;
typedef __bf16 v16bf __attribute__((ext_vector_type(16)));
typedef __bf16 v2bf  __attribute__((ext_vector_type(2)));
typedef float  v8f   __attribute__((ext_vector_type(8)));

// Monotone order-preserving float <-> uint key (for atomicMax on floats).
__device__ __forceinline__ unsigned f32_key(float f) {
  unsigned b = __float_as_uint(f);
  return (b & 0x80000000u) ? ~b : (b | 0x80000000u);
}
__device__ __forceinline__ float key_f32(unsigned k) {
  unsigned b = (k & 0x80000000u) ? (k & 0x7fffffffu) : ~k;
  return __uint_as_float(b);
}

// ---------------------------------------------------------------------------
// WMMA GEMM: out[N,NC] = epilogue( A[N,64] @ W[64,NC] )
//   epilogue: +bias, optional BN(eval), optional +addsrc, optional ReLU
// 128 threads = 4 waves; wave w computes rows [blk*64 + 16w, +16) x NC.
// A, W converted to bf16 in LDS; K=64 done as two 16x16x32 bf16 WMMAs.
// ---------------------------------------------------------------------------
template <int NC, bool BN, bool RELU, bool ADD>
__global__ __launch_bounds__(128)
void gemm_wmma_kernel(const float* __restrict__ A, int lda,
                      const float* __restrict__ W,      // [64,NC] row-major
                      const float* __restrict__ bias,   // [NC] or nullptr
                      const float* __restrict__ gamma,
                      const float* __restrict__ beta,
                      const float* __restrict__ mean,
                      const float* __restrict__ var,
                      const float* __restrict__ addsrc, int ldadd,
                      float* __restrict__ out, int ldo,
                      int n)
{
  __shared__ bf16_t sWt[NC][66];    // W transposed, padded (bank-conflict free)
  __shared__ float  sA[4][16][68];  // per-wave A tile, padded

  const int tid  = threadIdx.x;
  const int lane = tid & 31;
  const int wv   = tid >> 5;
  const int half = lane >> 4;       // 0: lanes 0-15, 1: lanes 16-31
  const int m    = lane & 15;

  // Stage W (f32 -> bf16, transposed)
  for (int idx = tid; idx < 64 * NC; idx += 128) {
    int k = idx / NC, c = idx - k * NC;
    sWt[c][k] = (bf16_t)W[idx];
  }

  // Stage this wave's 16x64 A tile (coalesced float4 loads, rows clamped)
  const int rowBase = blockIdx.x * 64 + wv * 16;
  #pragma unroll
  for (int j = 0; j < 8; ++j) {
    int flat = j * 32 + lane;              // 256 float4 = 16 rows x 16
    int rr = flat >> 4, cc = (flat & 15) << 2;
    int grow = rowBase + rr; if (grow > n - 1) grow = n - 1;
    const float4 val = *(const float4*)(A + (size_t)grow * lda + cc);
    *(float4*)&sA[wv][rr][cc] = val;
  }
  __syncthreads();

  // Build A fragments (ISA 7.12.2: 16-bit A 16x32; VGPR v holds K pair
  //   base = v<4 ? 2v + 8*half : 16 + 2(v-4) + 8*half)
  v16bf a0, a1;
  #pragma unroll
  for (int i = 0; i < 8; ++i) {
    int kk = (i < 4) ? (2 * i + 8 * half) : (16 + 2 * (i - 4) + 8 * half);
    float2 f0 = *(const float2*)&sA[wv][m][kk];
    float2 f1 = *(const float2*)&sA[wv][m][kk + 32];
    a0[2 * i] = (bf16_t)f0.x;  a0[2 * i + 1] = (bf16_t)f0.y;
    a1[2 * i] = (bf16_t)f1.x;  a1[2 * i + 1] = (bf16_t)f1.y;
  }

  const bool fullTile = (rowBase + 16 <= n);   // wave-uniform tail check

  // Column tiles of 16
  #pragma unroll
  for (int t = 0; t < NC / 16; ++t) {
    const int col = t * 16 + m;

    // B fragment (32x16): VGPR j holds K pair base = j<4 ? 2j + 8*half : 16 + 2(j-4) + 8*half, N = lane%16
    v16bf b0, b1;
    #pragma unroll
    for (int j = 0; j < 8; ++j) {
      int k0 = (j < 4) ? (2 * j + 8 * half) : (16 + 2 * (j - 4) + 8 * half);
      v2bf p0 = *(const v2bf*)&sWt[col][k0];
      v2bf p1 = *(const v2bf*)&sWt[col][k0 + 32];
      b0[2 * j] = p0.x;  b0[2 * j + 1] = p0.y;
      b1[2 * j] = p1.x;  b1[2 * j + 1] = p1.y;
    }

    v8f acc = {};
    acc = __builtin_amdgcn_wmma_f32_16x16x32_bf16(false, a0, false, b0,
                                                  (short)0, acc, false, false);
    acc = __builtin_amdgcn_wmma_f32_16x16x32_bf16(false, a1, false, b1,
                                                  (short)0, acc, false, false);

    // Epilogue (column-uniform params loaded once)
    const float bv = bias ? bias[col] : 0.0f;
    float g = 0.f, bt = 0.f, mu = 0.f, rv = 0.f;
    if (BN) { g = gamma[col]; bt = beta[col]; mu = mean[col];
              rv = rsqrtf(var[col] + 1e-5f); }

    // C/D layout: VGPR r -> M = r + 8*half, N = lane%16
    if (fullTile) {
      #pragma unroll
      for (int r = 0; r < 8; ++r) {
        const int row = rowBase + r + 8 * half;
        float v0 = acc[r] + bv;
        if (BN)   v0 = (v0 - mu) * rv * g + bt;
        if (ADD)  v0 += addsrc[(size_t)row * ldadd + col];
        if (RELU) v0 = fmaxf(v0, 0.0f);
        out[(size_t)row * ldo + col] = v0;
      }
    } else {
      #pragma unroll
      for (int r = 0; r < 8; ++r) {
        const int row = rowBase + r + 8 * half;
        if (row < n) {
          float v0 = acc[r] + bv;
          if (BN)   v0 = (v0 - mu) * rv * g + bt;
          if (ADD)  v0 += addsrc[(size_t)row * ldadd + col];
          if (RELU) v0 = fmaxf(v0, 0.0f);
          out[(size_t)row * ldo + col] = v0;
        }
      }
    }
  }
}

// ---------------------------------------------------------------------------
// Edge / elementwise kernels (L2-resident gathers + f32 atomics)
// ---------------------------------------------------------------------------
__global__ void copy_slice_kernel(const float* __restrict__ src, int lds,
                                  float* __restrict__ dst, int ldd, int n) {
  int t = blockIdx.x * blockDim.x + threadIdx.x;
  if (t >= n * 16) return;
  int i = t >> 4, c = (t & 15) << 2;
  *(float4*)(dst + (size_t)i * ldd + c) =
      *(const float4*)(src + (size_t)i * lds + c);
}

__global__ void deg_kernel(const int* __restrict__ dst, float* __restrict__ deg, int E) {
  int e = blockIdx.x * blockDim.x + threadIdx.x;
  if (e < E) atomicAdd(deg + dst[e], 1.0f);
}

// Grid-stride with software prefetch of next iteration's gather row
// (random rows of a 25.6MB feature map -> L2-resident; global_prefetch_b8
//  hides the L2 hit latency behind the current iteration's atomics).
__global__ void seg_add_kernel(const float* __restrict__ h, int lda,
                               const int* __restrict__ src,
                               const int* __restrict__ dst,
                               float* __restrict__ out, int E) {
  const int total  = E * 16;
  const int stride = (int)(gridDim.x * blockDim.x);
  for (int t = blockIdx.x * blockDim.x + threadIdx.x; t < total; t += stride) {
    const int tn = t + stride;
    if (tn < total) {
      const int en = tn >> 4, cn = (tn & 15) << 2;
      __builtin_prefetch(h + (size_t)src[en] * lda + cn, 0, 3);
    }
    const int e = t >> 4, c = (t & 15) << 2;
    const int s = src[e], d = dst[e];
    const float4 hv = *(const float4*)(h + (size_t)s * lda + c);
    float* o = out + (size_t)d * 64 + c;
    atomicAdd(o + 0, hv.x); atomicAdd(o + 1, hv.y);
    atomicAdd(o + 2, hv.z); atomicAdd(o + 3, hv.w);
  }
}

__global__ void sage_mean_kernel(const float* __restrict__ agg,
                                 const float* __restrict__ deg,
                                 float* __restrict__ out, int n) {
  int t = blockIdx.x * blockDim.x + threadIdx.x;
  if (t >= n * 16) return;
  int i = t >> 4, c = (t & 15) << 2;
  float inv = 1.0f / fmaxf(deg[i], 1.0f);
  float4 a = *(const float4*)(agg + (size_t)i * 64 + c);
  a.x *= inv; a.y *= inv; a.z *= inv; a.w *= inv;
  *(float4*)(out + (size_t)i * 64 + c) = a;
}

__global__ void attn_score_kernel(const float* __restrict__ q,
                                  const float* __restrict__ k,
                                  const int* __restrict__ src,
                                  const int* __restrict__ dst,
                                  float* __restrict__ score,
                                  unsigned* __restrict__ smax, int E) {
  int t = blockIdx.x * blockDim.x + threadIdx.x;
  if (t >= 2 * E) return;
  int e = t >> 1, hh = t & 1;
  const float4* qp = (const float4*)(q + (size_t)dst[e] * 128 + hh * 64);
  const float4* kp = (const float4*)(k + (size_t)src[e] * 128 + hh * 64);
  float acc = 0.0f;
  #pragma unroll
  for (int i = 0; i < 16; ++i) {
    float4 a = qp[i], b = kp[i];
    acc += a.x * b.x + a.y * b.y + a.z * b.z + a.w * b.w;
  }
  acc *= 0.125f;                                   // 1/sqrt(64)
  score[t] = acc;
  atomicMax(smax + (size_t)dst[e] * 2 + hh, f32_key(acc));
}

__global__ void attn_exp_kernel(float* __restrict__ score,
                                const unsigned* __restrict__ smax,
                                float* __restrict__ denom,
                                const int* __restrict__ dst, int E) {
  int t = blockIdx.x * blockDim.x + threadIdx.x;
  if (t >= 2 * E) return;
  int e = t >> 1, hh = t & 1;
  float mx = key_f32(smax[(size_t)dst[e] * 2 + hh]);
  float ev = __expf(score[t] - mx);
  score[t] = ev;
  atomicAdd(denom + (size_t)dst[e] * 2 + hh, ev);
}

__global__ void attn_agg_kernel(const float* __restrict__ escore,
                                const float* __restrict__ denom,
                                const float* __restrict__ v,
                                const int* __restrict__ src,
                                const int* __restrict__ dst,
                                float* __restrict__ out, int E) {
  int t = blockIdx.x * blockDim.x + threadIdx.x;
  if (t >= E * 32) return;
  int e = t >> 5, r = t & 31, hh = r >> 4, c = (r & 15) << 2;
  int s = src[e], d = dst[e];
  float alpha = escore[e * 2 + hh] / denom[(size_t)d * 2 + hh] * 0.5f; // head mean
  const float4 vv = *(const float4*)(v + (size_t)s * 128 + hh * 64 + c);
  float* o = out + (size_t)d * 64 + c;
  atomicAdd(o + 0, alpha * vv.x); atomicAdd(o + 1, alpha * vv.y);
  atomicAdd(o + 2, alpha * vv.z); atomicAdd(o + 3, alpha * vv.w);
}

__global__ void cls_kernel(const float* __restrict__ xs,
                           const float* __restrict__ w,   // [384,2]
                           const float* __restrict__ b,   // [2]
                           float* __restrict__ out, int n) {
  int i = blockIdx.x * blockDim.x + threadIdx.x;
  if (i >= n) return;
  const float4* xr = (const float4*)(xs + (size_t)i * 384);
  const float2* wp = (const float2*)w;
  float a0 = b[0], a1 = b[1];
  #pragma unroll 4
  for (int j = 0; j < 96; ++j) {
    float4 xv = xr[j];
    float2 w0 = wp[j * 4 + 0], w1 = wp[j * 4 + 1];
    float2 w2 = wp[j * 4 + 2], w3 = wp[j * 4 + 3];
    a0 += xv.x * w0.x + xv.y * w1.x + xv.z * w2.x + xv.w * w3.x;
    a1 += xv.x * w0.y + xv.y * w1.y + xv.z * w2.y + xv.w * w3.y;
  }
  out[(size_t)i * 2 + 0] = a0;
  out[(size_t)i * 2 + 1] = a1;
}

// ---------------------------------------------------------------------------
// Launcher
// ---------------------------------------------------------------------------
extern "C" void kernel_launch(void* const* d_in, const int* in_sizes, int n_in,
                              void* d_out, int out_size, void* d_ws, size_t ws_size,
                              hipStream_t stream) {
  const int n = in_sizes[0] / 64;     // 100000
  const int e = in_sizes[1] / 2;      // 1000000

  const float* x   = (const float*)d_in[0];
  const int*   ei  = (const int*)d_in[1];
  const int*   srcp = ei;
  const int*   dstp = ei + e;
  auto F = [&](int i) { return (const float*)d_in[i]; };

  // Scratch carve (256B aligned)
  char* wsp = (char*)d_ws;
  auto carve = [&](size_t bytes) -> void* {
    void* p = (void*)wsp; wsp += (bytes + 255) & ~(size_t)255; return p;
  };
  float*    xs    = (float*)carve((size_t)n * 384 * sizeof(float));
  float*    agg   = (float*)carve((size_t)n * 64  * sizeof(float));
  float*    tmp   = (float*)carve((size_t)n * 64  * sizeof(float));
  float*    qb    = (float*)carve((size_t)n * 128 * sizeof(float));
  float*    kb    = (float*)carve((size_t)n * 128 * sizeof(float));
  float*    vb    = (float*)carve((size_t)n * 128 * sizeof(float));
  float*    esc   = (float*)carve((size_t)e * 2   * sizeof(float));
  float*    denom = (float*)carve((size_t)n * 2   * sizeof(float));
  unsigned* smax  = (unsigned*)carve((size_t)n * 2 * sizeof(unsigned));
  float*    deg   = (float*)carve((size_t)n * sizeof(float));

  const int TPB = 256;
  auto nb = [&](long long total) { return (unsigned)((total + TPB - 1) / TPB); };
  const unsigned gemmBlocks = (unsigned)((n + 63) / 64);
  const unsigned segBlocks  = 4096;   // grid-stride (prefetch pipelined)

  // Degree (same for all SAGE layers) — compute once per launch
  hipMemsetAsync(deg, 0, (size_t)n * sizeof(float), stream);
  deg_kernel<<<nb(e), TPB, 0, stream>>>(dstp, deg, e);

  auto run_gin = [&](const float* h, int ldh, int p0, float* xout) {
    // agg = h + seg(h[src])
    copy_slice_kernel<<<nb((long long)n * 16), TPB, 0, stream>>>(h, ldh, agg, 64, n);
    seg_add_kernel<<<segBlocks, TPB, 0, stream>>>(h, ldh, srcp, dstp, agg, e);
    // tmp = relu(BN(agg @ w1 + b1))
    gemm_wmma_kernel<64, true, true, false><<<gemmBlocks, 128, 0, stream>>>(
        agg, 64, F(p0 + 0), F(p0 + 1), F(p0 + 2), F(p0 + 3), F(p0 + 4), F(p0 + 5),
        nullptr, 0, tmp, 64, n);
    // xout = relu(tmp @ w2 + b2)   (outer layer ReLU)
    gemm_wmma_kernel<64, false, true, false><<<gemmBlocks, 128, 0, stream>>>(
        tmp, 64, F(p0 + 6), F(p0 + 7), nullptr, nullptr, nullptr, nullptr,
        nullptr, 0, xout, 384, n);
  };

  auto run_sage = [&](const float* h, int ldh, int p0, float* xout) {
    hipMemsetAsync(agg, 0, (size_t)n * 64 * sizeof(float), stream);
    seg_add_kernel<<<segBlocks, TPB, 0, stream>>>(h, ldh, srcp, dstp, agg, e);
    sage_mean_kernel<<<nb((long long)n * 16), TPB, 0, stream>>>(agg, deg, tmp, n);
    // agg = mean @ wl + bl
    gemm_wmma_kernel<64, false, false, false><<<gemmBlocks, 128, 0, stream>>>(
        tmp, 64, F(p0 + 0), F(p0 + 1), nullptr, nullptr, nullptr, nullptr,
        nullptr, 0, agg, 64, n);
    // xout = relu(h @ wr + agg)
    gemm_wmma_kernel<64, false, true, true><<<gemmBlocks, 128, 0, stream>>>(
        h, ldh, F(p0 + 2), nullptr, nullptr, nullptr, nullptr, nullptr,
        agg, 64, xout, 384, n);
  };

  auto run_tr = [&](const float* h, int ldh, int p0, float* xout) {
    gemm_wmma_kernel<128, false, false, false><<<gemmBlocks, 128, 0, stream>>>(
        h, ldh, F(p0 + 0), F(p0 + 1), nullptr, nullptr, nullptr, nullptr,
        nullptr, 0, qb, 128, n);
    gemm_wmma_kernel<128, false, false, false><<<gemmBlocks, 128, 0, stream>>>(
        h, ldh, F(p0 + 2), F(p0 + 3), nullptr, nullptr, nullptr, nullptr,
        nullptr, 0, kb, 128, n);
    gemm_wmma_kernel<128, false, false, false><<<gemmBlocks, 128, 0, stream>>>(
        h, ldh, F(p0 + 4), F(p0 + 5), nullptr, nullptr, nullptr, nullptr,
        nullptr, 0, vb, 128, n);
    hipMemsetAsync(smax,  0, (size_t)n * 2 * sizeof(unsigned), stream);
    hipMemsetAsync(denom, 0, (size_t)n * 2 * sizeof(float), stream);
    hipMemsetAsync(tmp,   0, (size_t)n * 64 * sizeof(float), stream);
    attn_score_kernel<<<nb((long long)e * 2), TPB, 0, stream>>>(qb, kb, srcp, dstp, esc, smax, e);
    attn_exp_kernel<<<nb((long long)e * 2), TPB, 0, stream>>>(esc, smax, denom, dstp, e);
    attn_agg_kernel<<<nb((long long)e * 32), TPB, 0, stream>>>(esc, denom, vb, srcp, dstp, tmp, e);
    // xout = relu(h @ ws + bs + attn_out)
    gemm_wmma_kernel<64, false, true, true><<<gemmBlocks, 128, 0, stream>>>(
        h, ldh, F(p0 + 6), F(p0 + 7), nullptr, nullptr, nullptr, nullptr,
        tmp, 64, xout, 384, n);
  };

  // Param indices (setup_inputs dict order, recursive insertion order):
  // l0 gin: 2..9 | l1 sage: 10..12 | l2 tr: 13..20
  // l3 gin: 21..28 | l4 sage: 29..31 | l5 tr: 32..39 | cls: 40,41
  run_gin (x,            64,  2,  xs + 0 * 64);
  run_sage(xs + 0 * 64, 384, 10,  xs + 1 * 64);
  run_tr  (xs + 1 * 64, 384, 13,  xs + 2 * 64);
  run_gin (xs + 2 * 64, 384, 21,  xs + 3 * 64);
  run_sage(xs + 3 * 64, 384, 29,  xs + 4 * 64);
  run_tr  (xs + 4 * 64, 384, 32,  xs + 5 * 64);

  cls_kernel<<<nb(n), TPB, 0, stream>>>(xs, F(40), F(41), (float*)d_out, n);
}
